// VisionTritonAttention_60627758350370
// MI455X (gfx1250) — matrix operand order repaired and checked by reading, compiled, not run on hardware
//
#include <hip/hip_runtime.h>

typedef _Float16 half16 __attribute__((ext_vector_type(16)));
typedef _Float16 half8  __attribute__((ext_vector_type(8)));
typedef __fp16   fp16x2 __attribute__((ext_vector_type(2)));
typedef float    float8 __attribute__((ext_vector_type(8)));
typedef float    float4v __attribute__((ext_vector_type(4)));
typedef int      int8v  __attribute__((ext_vector_type(8)));
typedef int      int2v  __attribute__((ext_vector_type(2)));

#define N_HEADS    16
#define HEAD_DIM   64
#define ROW_STRIDE (N_HEADS * HEAD_DIM)   // floats per token row
#define KVTILE     32                     // kv rows per iteration
#define QW         32                     // q rows per wave (2 x 16 subtiles)
#define QTILE      128                    // q rows per block (4 waves)
#define MAX_QTILES 8                      // ceil(max_seqlen=1024 / 128)
#define NTHREADS   128
#define LOG2E      1.4426950408889634f

static __device__ inline float8 wmma16(half16 a, half16 b, float8 c) {
    return __builtin_amdgcn_wmma_f32_16x16x32_f16(false, a, false, b, (short)0, c, false, false);
}

__global__ __launch_bounds__(NTHREADS)
void varlen_attn_wmma(const float* __restrict__ q,
                      const float* __restrict__ k,
                      const float* __restrict__ v,
                      const int*   __restrict__ cu,
                      float*       __restrict__ out)
{
    alignas(16) __shared__ _Float16 sK[KVTILE * HEAD_DIM];   // (kv, dim) row-major
    alignas(16) __shared__ _Float16 sV[HEAD_DIM * KVTILE];   // (dim, kv) transposed

    const int bid  = blockIdx.x;
    const int qt   = bid & (MAX_QTILES - 1);
    const int head = (bid >> 3) & (N_HEADS - 1);
    const int b    = bid >> 7;

    const int start = cu[b];
    const int L     = cu[b + 1] - start;
    if (qt * QTILE >= L) return;              // uniform exit before any barrier

    const int tid  = threadIdx.x;
    const int lane = tid & 31;
    const int wave = tid >> 5;
    const int n    = lane & 15;
    const int hi   = lane >> 4;

    // ---- Q^T B-fragments for 2 subtiles, pre-scaled by log2(e)/sqrt(64) ----
    // B-layout (32x16 f16): lane = col N=n, element e <-> K = 8*hi + e (e<8), 16 + 8*hi + (e-8) (e>=8).
    const float qscale = 0.125f * LOG2E;
    int   q_local[2];
    half16 bq[2][2];
    #pragma unroll
    for (int s = 0; s < 2; ++s) {
        q_local[s] = qt * QTILE + wave * QW + s * 16 + n;
        const int qc = (q_local[s] < L) ? q_local[s] : (L - 1);
        const float* qp = q + (size_t)(start + qc) * ROW_STRIDE + head * HEAD_DIM;
        const int d0 = 8 * hi;
        #pragma unroll
        for (int e = 0; e < 16; ++e) {
            bq[s][0][e] = (_Float16)(qp[d0 + e + (e >= 8 ? 8 : 0)]      * qscale);
            bq[s][1][e] = (_Float16)(qp[32 + d0 + e + (e >= 8 ? 8 : 0)] * qscale);
        }
    }

    float8 ot[2][4] = {{{}, {}, {}, {}}, {{}, {}, {}, {}}};
    float  m_run[2] = {-3.0e38f, -3.0e38f};
    float  l_run[2] = {0.0f, 0.0f};

    // 16-bit A-fragment loader: lane row = `row`; element e:
    //   e<8: K = 8*hi + e ; e>=8: K = 16 + 8*hi + (e-8)  -> two aligned b128 runs
    auto ldA = [&](const _Float16* base, int row, int stride) -> half16 {
        const _Float16* r = base + row * stride;
        half8 lo = *reinterpret_cast<const half8*>(r + 8 * hi);
        half8 hh = *reinterpret_cast<const half8*>(r + 16 + 8 * hi);
        return __builtin_shufflevector(lo, hh, 0,1,2,3,4,5,6,7,8,9,10,11,12,13,14,15);
    };

    // ---- one 32-key tile; `mask` is compile-time folded at both call sites ----
    auto tile = [&](int kv0, bool mask) {
        __syncthreads();                       // WAR guard on LDS
        #pragma unroll
        for (int i = 0; i < 4; ++i) {          // 128 thr * 4 iters * float4 = 2048 elts
            const int base = i * 512 + tid * 4;
            const int kv = base >> 6, dim = base & 63;
            int row = kv0 + kv;
            if (mask) row = (row < L) ? row : (L - 1);
            const size_t off = (size_t)(start + row) * ROW_STRIDE + head * HEAD_DIM + dim;
            const float4v kf = *reinterpret_cast<const float4v*>(k + off);
            const float4v vf = *reinterpret_cast<const float4v*>(v + off);
            const fp16x2 k01 = __builtin_amdgcn_cvt_pkrtz(kf[0], kf[1]);
            const fp16x2 k23 = __builtin_amdgcn_cvt_pkrtz(kf[2], kf[3]);
            const int2v kh = { __builtin_bit_cast(int, k01),
                               __builtin_bit_cast(int, k23) };
            *reinterpret_cast<int2v*>(sK + base) = kh;      // ds_store_b64
            sV[(dim + 0) * KVTILE + kv] = (_Float16)vf[0];
            sV[(dim + 1) * KVTILE + kv] = (_Float16)vf[1];
            sV[(dim + 2) * KVTILE + kv] = (_Float16)vf[2];
            sV[(dim + 3) * KVTILE + kv] = (_Float16)vf[3];
        }
        __syncthreads();

        // K A-fragments (shared by both q subtiles) and V^T A-fragments
        const half16 ak00 = ldA(sK, n,      HEAD_DIM);
        const half16 ak01 = ldA(sK + 32, n, HEAD_DIM);
        const half16 ak10 = ldA(sK, n + 16,      HEAD_DIM);
        const half16 ak11 = ldA(sK + 32, n + 16, HEAD_DIM);
        half16 av[4];
        #pragma unroll
        for (int d = 0; d < 4; ++d) av[d] = ldA(sV, 16 * d + n, KVTILE);

        const int kb0 = kv0 + 8 * hi;          // st0[j] <-> kv kb0+j
        const int kb1 = kv0 + 16 + 8 * hi;     // st1[j] <-> kv kb1+j

        #pragma unroll
        for (int s = 0; s < 2; ++s) {
            // S^T = K_tile x Q^T (log2-domain scores), two C-tiles
            float8 st0 = wmma16(ak00, bq[s][0], (float8){});
            st0        = wmma16(ak01, bq[s][1], st0);
            float8 st1 = wmma16(ak10, bq[s][0], (float8){});
            st1        = wmma16(ak11, bq[s][1], st1);

            // online softmax (lane <-> q col; single cross-half shuffle)
            float p0[8], p1[8], mloc = -3.0e38f;
            #pragma unroll
            for (int j = 0; j < 8; ++j) {
                float s0 = st0[j], s1 = st1[j];
                if (mask) {
                    s0 = (kb0 + j < L) ? s0 : -1.0e9f;
                    s1 = (kb1 + j < L) ? s1 : -1.0e9f;
                }
                p0[j] = s0; p1[j] = s1;
                mloc = fmaxf(mloc, fmaxf(s0, s1));
            }
            mloc = fmaxf(mloc, __shfl_xor(mloc, 16, 32));
            const float m_new = fmaxf(m_run[s], mloc);
            const float alpha = __builtin_amdgcn_exp2f(m_run[s] - m_new);
            float rsum = 0.0f;
            #pragma unroll
            for (int j = 0; j < 8; ++j) {
                p0[j] = __builtin_amdgcn_exp2f(p0[j] - m_new);
                p1[j] = __builtin_amdgcn_exp2f(p1[j] - m_new);
                rsum += p0[j] + p1[j];
            }
            rsum += __shfl_xor(rsum, 16, 32);
            l_run[s] = l_run[s] * alpha + rsum;
            m_run[s] = m_new;
            #pragma unroll
            for (int j = 0; j < 8; ++j) {
                ot[s][0][j] *= alpha; ot[s][1][j] *= alpha;
                ot[s][2][j] *= alpha; ot[s][3][j] *= alpha;
            }

            // P^T C-layout -> B-fragment (K = 8*hi + e for e<8, 16 + 8*hi + (e-8) for e>=8):
            // pack 2 probs per dword, own lane's st0 then st1.
            int pk0[4], pk1[4];
            #pragma unroll
            for (int jj = 0; jj < 4; ++jj) {
                pk0[jj] = __builtin_bit_cast(int,
                    __builtin_amdgcn_cvt_pkrtz(p0[2*jj], p0[2*jj + 1]));
                pk1[jj] = __builtin_bit_cast(int,
                    __builtin_amdgcn_cvt_pkrtz(p1[2*jj], p1[2*jj + 1]));
            }
            int8v bpi;
            #pragma unroll
            for (int jj = 0; jj < 4; ++jj) {
                bpi[jj]     = pk0[jj];
                bpi[jj + 4] = pk1[jj];
            }
            const half16 bp = __builtin_bit_cast(half16, bpi);

            // O^T += V^T x P^T
            #pragma unroll
            for (int d = 0; d < 4; ++d)
                ot[s][d] = wmma16(av[d], bp, ot[s][d]);
        }
    };

    const int nfull = L >> 5;                  // tiles with kv0+32 <= L
    for (int it = 0; it < nfull; ++it) {
        if (it + 1 < nfull) {                  // gfx1250 global_prefetch_b8, next tile
            int prow = (it + 1) * KVTILE + (tid >> 2);
            prow = (prow < L) ? prow : (L - 1);
            const size_t poff = (size_t)(start + prow) * ROW_STRIDE
                              + head * HEAD_DIM + (tid & 3) * 16;
            __builtin_prefetch(k + poff, 0, 1);
            __builtin_prefetch(v + poff, 0, 1);
        }
        tile(it * KVTILE, false);
    }
    if (L & (KVTILE - 1)) tile(nfull * KVTILE, true);

    // ---- epilogue ----
    #pragma unroll
    for (int s = 0; s < 2; ++s) {
        if (q_local[s] < L) {
            const float inv = 1.0f / l_run[s];
            float* orow = out + (size_t)(start + q_local[s]) * ROW_STRIDE
                        + head * HEAD_DIM + 8 * hi;
            #pragma unroll
            for (int d = 0; d < 4; ++d) {
                float4v lo = { ot[s][d][0] * inv, ot[s][d][1] * inv,
                               ot[s][d][2] * inv, ot[s][d][3] * inv };
                float4v hh = { ot[s][d][4] * inv, ot[s][d][5] * inv,
                               ot[s][d][6] * inv, ot[s][d][7] * inv };
                *reinterpret_cast<float4v*>(orow + 16 * d)     = lo;
                *reinterpret_cast<float4v*>(orow + 16 * d + 4) = hh;
            }
        }
    }
}

extern "C" void kernel_launch(void* const* d_in, const int* in_sizes, int n_in,
                              void* d_out, int out_size, void* d_ws, size_t ws_size,
                              hipStream_t stream) {
    const float* q  = (const float*)d_in[0];
    const float* k  = (const float*)d_in[1];
    const float* v  = (const float*)d_in[2];
    const int*   cu = (const int*)d_in[4];
    float* out = (float*)d_out;

    const int n_batch = in_sizes[4] - 1;
    const int grid = n_batch * N_HEADS * MAX_QTILES;

    varlen_attn_wmma<<<dim3(grid), dim3(NTHREADS), 0, stream>>>(q, k, v, cu, out);
}
